// SIGN_learnable_emb_Module_46608985096736
// MI455X (gfx1250) — matrix operand order repaired and checked by reading, not compiled, not run on hardware
//
#include <hip/hip_runtime.h>
#include <hip/hip_bf16.h>

// Problem constants (from the reference)
#define N0 1000000
#define N1 100000
#define N2 10000
#define E0 1000000
#define E1 100000
#define DIM 128
#define TD  (3 * DIM)   // 384

typedef __attribute__((ext_vector_type(16))) __bf16 v16bf;
typedef __attribute__((ext_vector_type(8)))  float  v8f;

// ---------------------------------------------------------------------------
// CSR build: histogram of destinations
// ---------------------------------------------------------------------------
__global__ void hist_kernel(const int* __restrict__ dst, int* __restrict__ cnt, int n) {
    int i = blockIdx.x * blockDim.x + threadIdx.x;
    if (i < n) atomicAdd(&cnt[dst[i]], 1);
}

// Single-workgroup exclusive prefix scan over up to ~100K bins.
// Writes exclusive sums to both row_start and pos (pos is consumed by scatter).
__global__ __launch_bounds__(1024) void scan_kernel(const int* __restrict__ cnt,
                                                    int* __restrict__ row_start,
                                                    int* __restrict__ pos, int n) {
    __shared__ int part[1024];
    const int tid = threadIdx.x;
    const int chunk = (n + 1023) / 1024;
    const int begin = tid * chunk;
    const int end = min(begin + chunk, n);

    int s = 0;
    for (int i = begin; i < end; ++i) s += cnt[i];
    part[tid] = s;
    __syncthreads();

    // Hillis–Steele inclusive scan of the 1024 partials
    for (int off = 1; off < 1024; off <<= 1) {
        int add = 0;
        if (tid >= off) add = part[tid - off];
        __syncthreads();
        part[tid] += add;
        __syncthreads();
    }

    int base = (tid == 0) ? 0 : part[tid - 1];
    for (int i = begin; i < end; ++i) {
        row_start[i] = base;
        pos[i] = base;
        base += cnt[i];
    }
}

// Scatter edge ids into CSR order (pos holds running insert cursors)
__global__ void scatter_kernel(const int* __restrict__ dst, int* __restrict__ pos,
                               int* __restrict__ perm, int n) {
    int i = blockIdx.x * blockDim.x + threadIdx.x;
    if (i < n) {
        int p = atomicAdd(&pos[dst[i]], 1);
        perm[p] = i;
    }
}

// ---------------------------------------------------------------------------
// SpMM gather: one wave32 per destination row; lane l owns features [4l,4l+4).
// 512B row read = one fully-coalesced global_load_b128 per lane. No atomics.
// ---------------------------------------------------------------------------
__global__ __launch_bounds__(256) void spmm_gather_kernel(
    const float* __restrict__ x, const int* __restrict__ src,
    const float* __restrict__ ew, const int* __restrict__ perm,
    const int* __restrict__ row_start, const int* __restrict__ cnt,
    float* __restrict__ out, int nrows) {
    const int gtid = blockIdx.x * blockDim.x + threadIdx.x;
    const int wave = gtid >> 5;
    const int lane = threadIdx.x & 31;
    if (wave >= nrows) return;

    const int beg = row_start[wave];
    const int n   = cnt[wave];
    const float4* __restrict__ x4 = (const float4*)x;

    float4 acc = make_float4(0.f, 0.f, 0.f, 0.f);
    int e = (n > 0) ? perm[beg] : 0;
    for (int j = 0; j < n; ++j) {
        const int s = src[e];
        const float w = ew[e];
        if (j + 1 < n) {
            e = perm[beg + j + 1];
            // hide gather latency: prefetch next source row (global_prefetch_b8)
            __builtin_prefetch(&x4[(size_t)src[e] * 32 + lane], 0, 1);
        }
        const float4 v = x4[(size_t)s * 32 + lane];
        acc.x += w * v.x; acc.y += w * v.y; acc.z += w * v.z; acc.w += w * v.w;
    }
    ((float4*)out)[(size_t)wave * 32 + lane] = acc;
}

// ---------------------------------------------------------------------------
// Fused MLP: one wave32 per 16-row tile.
//   A = [x | h1 | h2] tile (16x384) staged bf16 in LDS
//   hid = tanh(A @ W1 + b1)  (16x128, bf16 in LDS)
//   out = hid @ W2 + b2      (f32)
// WMMA fragment layouts follow the CDNA5 ISA tables (wave32).
// ---------------------------------------------------------------------------
__global__ __launch_bounds__(32) void mlp_kernel(
    const float* __restrict__ x,  const float* __restrict__ h1,
    const float* __restrict__ h2, const float* __restrict__ W1,
    const float* __restrict__ b1, const float* __restrict__ W2,
    const float* __restrict__ b2, float* __restrict__ out) {
    constexpr int AP = TD + 8;    // padded LDS stride for A tile (bf16 elems)
    constexpr int HP = DIM + 8;   // padded LDS stride for hid tile
    __shared__ __bf16 Alds[16 * AP];
    __shared__ __bf16 Hlds[16 * HP];

    const int tid = threadIdx.x;        // 0..31
    const int r0  = blockIdx.x * 16;    // output row base

    // ---- Stage concatenated feature tile as bf16 ----
    for (int idx = tid; idx < 16 * TD; idx += 32) {
        const int m = idx / TD, c = idx % TD;
        const int row = r0 + m;
        float v = 0.f;
        if (row < N2) {
            if (c < DIM)           v = x [(size_t)row * DIM + c];
            else if (c < 2 * DIM)  v = h1[(size_t)row * DIM + (c - DIM)];
            else                   v = h2[(size_t)row * DIM + (c - 2 * DIM)];
        }
        Alds[m * AP + c] = (__bf16)v;
    }
    __syncthreads();

    // Per-lane fragment geometry (ISA 7.12.2, wave32)
    const int  n16  = tid & 15;
    const bool hi   = (tid >= 16);
    const int  akb0 = hi ? 8  : 0;    // A elems 0..7 : K base
    const int  akb1 = hi ? 24 : 16;   // A elems 8..15: K base
    const int  bkb  = hi ? 8  : 0;    // B elems 0..7: K base (elems 8..15: +16)
    const int  mrow = n16;            // A fragment row for this lane
    const int  mb   = hi ? 8 : 0;     // C/D row base

    // ---- GEMM1: [16x384] @ [384x128] ----
    v8f acc[8] = {};
    for (int kk = 0; kk < TD / 32; ++kk) {
        const int k0 = kk * 32;
        v16bf a;
        #pragma unroll
        for (int i = 0; i < 8; ++i) a[i]     = Alds[mrow * AP + k0 + akb0 + i];
        #pragma unroll
        for (int i = 0; i < 8; ++i) a[8 + i] = Alds[mrow * AP + k0 + akb1 + i];
        #pragma unroll
        for (int nn = 0; nn < 8; ++nn) {
            const int n = nn * 16 + n16;
            v16bf b;
            #pragma unroll
            for (int i = 0; i < 16; ++i)
                b[i] = (__bf16)W1[(size_t)(k0 + bkb + i + (i >= 8 ? 8 : 0)) * DIM + n];
            acc[nn] = __builtin_amdgcn_wmma_f32_16x16x32_bf16(
                false, a, false, b, (short)0, acc[nn], false, false);
        }
    }

    // ---- bias + tanh -> hid tile (bf16, LDS) ----
    #pragma unroll
    for (int nn = 0; nn < 8; ++nn) {
        const int n = nn * 16 + n16;
        const float bias = b1[n];
        #pragma unroll
        for (int j = 0; j < 8; ++j) {
            const int m = mb + j;
            Hlds[m * HP + n] = (__bf16)tanhf(acc[nn][j] + bias);
        }
    }
    __syncthreads();

    // ---- GEMM2: [16x128] @ [128x128] ----
    v8f acc2[8] = {};
    for (int kk = 0; kk < DIM / 32; ++kk) {
        const int k0 = kk * 32;
        v16bf a;
        #pragma unroll
        for (int i = 0; i < 8; ++i) a[i]     = Hlds[mrow * HP + k0 + akb0 + i];
        #pragma unroll
        for (int i = 0; i < 8; ++i) a[8 + i] = Hlds[mrow * HP + k0 + akb1 + i];
        #pragma unroll
        for (int nn = 0; nn < 8; ++nn) {
            const int n = nn * 16 + n16;
            v16bf b;
            #pragma unroll
            for (int i = 0; i < 16; ++i)
                b[i] = (__bf16)W2[(size_t)(k0 + bkb + i + (i >= 8 ? 8 : 0)) * DIM + n];
            acc2[nn] = __builtin_amdgcn_wmma_f32_16x16x32_bf16(
                false, a, false, b, (short)0, acc2[nn], false, false);
        }
    }

    // ---- bias + store ----
    #pragma unroll
    for (int nn = 0; nn < 8; ++nn) {
        const int n = nn * 16 + n16;
        const float bias = b2[n];
        #pragma unroll
        for (int j = 0; j < 8; ++j) {
            const int row = r0 + mb + j;
            if (row < N2) out[(size_t)row * DIM + n] = acc2[nn][j] + bias;
        }
    }
}

// ---------------------------------------------------------------------------
// Launcher
// ---------------------------------------------------------------------------
static inline char* ws_take(char*& p, size_t bytes) {
    char* r = p;
    p += (bytes + 255) & ~(size_t)255;
    return r;
}

extern "C" void kernel_launch(void* const* d_in, const int* in_sizes, int n_in,
                              void* d_out, int out_size, void* d_ws, size_t ws_size,
                              hipStream_t stream) {
    (void)in_sizes; (void)n_in; (void)out_size; (void)ws_size;

    const float* x    = (const float*)d_in[0];
    const int*   src0 = (const int*)  d_in[1];
    const int*   dst0 = (const int*)  d_in[2];
    const float* ew0  = (const float*)d_in[3];
    const int*   src1 = (const int*)  d_in[4];
    const int*   dst1 = (const int*)  d_in[5];
    const float* ew1  = (const float*)d_in[6];
    const float* W1   = (const float*)d_in[7];
    const float* b1   = (const float*)d_in[8];
    const float* W2   = (const float*)d_in[9];
    const float* b2   = (const float*)d_in[10];
    float* out = (float*)d_out;

    // Workspace carve-out (~62 MB)
    char* p = (char*)d_ws;
    float* h1    = (float*)ws_take(p, (size_t)N1 * DIM * sizeof(float));
    float* h2    = (float*)ws_take(p, (size_t)N2 * DIM * sizeof(float));
    int*   cnt0  = (int*)  ws_take(p, (size_t)N1 * sizeof(int));
    int*   rs0   = (int*)  ws_take(p, (size_t)N1 * sizeof(int));
    int*   pos0  = (int*)  ws_take(p, (size_t)N1 * sizeof(int));
    int*   perm0 = (int*)  ws_take(p, (size_t)E0 * sizeof(int));
    int*   cnt1  = (int*)  ws_take(p, (size_t)N2 * sizeof(int));
    int*   rs1   = (int*)  ws_take(p, (size_t)N2 * sizeof(int));
    int*   pos1  = (int*)  ws_take(p, (size_t)N2 * sizeof(int));
    int*   perm1 = (int*)  ws_take(p, (size_t)E1 * sizeof(int));

    // Zero CSR counters (capture-safe)
    hipMemsetAsync(cnt0, 0, (size_t)N1 * sizeof(int), stream);
    hipMemsetAsync(cnt1, 0, (size_t)N2 * sizeof(int), stream);

    // Build CSR for both hops
    hist_kernel<<<(E0 + 255) / 256, 256, 0, stream>>>(dst0, cnt0, E0);
    hist_kernel<<<(E1 + 255) / 256, 256, 0, stream>>>(dst1, cnt1, E1);
    scan_kernel<<<1, 1024, 0, stream>>>(cnt0, rs0, pos0, N1);
    scan_kernel<<<1, 1024, 0, stream>>>(cnt1, rs1, pos1, N2);
    scatter_kernel<<<(E0 + 255) / 256, 256, 0, stream>>>(dst0, pos0, perm0, E0);
    scatter_kernel<<<(E1 + 255) / 256, 256, 0, stream>>>(dst1, pos1, perm1, E1);

    // SpMM hops (one wave per destination row; 8 waves per 256-thread block)
    spmm_gather_kernel<<<(N1 + 7) / 8, 256, 0, stream>>>(
        x, src0, ew0, perm0, rs0, cnt0, h1, N1);
    spmm_gather_kernel<<<(N2 + 7) / 8, 256, 0, stream>>>(
        h1, src1, ew1, perm1, rs1, cnt1, h2, N2);

    // Fused MLP (WMMA bf16, one wave per 16-row tile)
    mlp_kernel<<<(N2 + 15) / 16, 32, 0, stream>>>(x, h1, h2, W1, b1, W2, b2, out);
}
